// QuantMSA_23948737643283
// MI455X (gfx1250) — compile-verified, matrix operand order checked by reading, not run on hardware
//
#include <hip/hip_runtime.h>
#include <cstdint>

// ---------------- problem constants ----------------
#define BB 64
#define SS 197
#define EE 768
#define HH 12
#define DD 64
#define MM (BB * SS)     // 12608 = 788*16 exactly
#define N1 (3 * EE)      // 2304
#define KK EE            // 768
#define ST 256           // padded attention row length (K for ctx WMMA)
#define EPSQ 1e-8f

typedef __attribute__((ext_vector_type(8))) int v8i;

// ---------------- helpers ----------------
__device__ __forceinline__ int kpat(int v, int half) {
  // 8-bit WMMA operand K pattern: dword v, lane-half h ->
  // K base = 16*(v>>1) + 4*(v&1) + 8*h   (CDNA5 ISA 16x64 8-bit layout)
  return ((v >> 1) << 4) | ((v & 1) << 2) | (half << 3);
}

__device__ __forceinline__ v8i load_frag(const int8_t* base, int half) {
  v8i f;
#pragma unroll
  for (int v = 0; v < 8; ++v)
    f[v] = *(const int*)(base + kpat(v, half));
  return f;
}

__device__ __forceinline__ float wave_fmax(float v) {
#pragma unroll
  for (int off = 16; off > 0; off >>= 1)
    v = fmaxf(v, __shfl_xor(v, off, 32));
  return v;
}

__device__ __forceinline__ float wave_fsum(float v) {
#pragma unroll
  for (int off = 16; off > 0; off >>= 1)
    v += __shfl_xor(v, off, 32);
  return v;
}

__device__ __forceinline__ float slot_scale(const unsigned* slot) {
  // scale = max(absmax/127, eps)
  return fmaxf(__uint_as_float(*slot) * (1.0f / 127.0f), EPSQ);
}

__device__ __forceinline__ int imin(int a, int b) { return a < b ? a : b; }

// ---------------- reduction / quant kernels ----------------
__global__ void zero_slots_kernel(unsigned* slots) {
  if (threadIdx.x < 16) slots[threadIdx.x] = 0u;
}

__global__ void absmax_kernel(const float* __restrict__ x, size_t n,
                              unsigned* __restrict__ slot) {
  size_t i = (size_t)blockIdx.x * blockDim.x + threadIdx.x;
  const size_t stride = (size_t)gridDim.x * blockDim.x;
  float m = 0.f;
  for (; i < n; i += stride) m = fmaxf(m, fabsf(x[i]));
  m = wave_fmax(m);
  if ((threadIdx.x & 31) == 0) atomicMax(slot, __float_as_uint(m));
}

__global__ void quantize_s8_kernel(const float* __restrict__ x, size_t n,
                                   const unsigned* __restrict__ slot,
                                   int8_t* __restrict__ q) {
  const float inv = 1.0f / slot_scale(slot);
  size_t i = (size_t)blockIdx.x * blockDim.x + threadIdx.x;
  const size_t stride = (size_t)gridDim.x * blockDim.x;
  for (; i < n; i += stride)
    q[i] = (int8_t)(int)fminf(127.f, fmaxf(-128.f, rintf(x[i] * inv)));
}

// quantize-dequantize in place, also produce absmax of the quantized tensor
__global__ void qdq_inplace_absmax_kernel(float* __restrict__ x, size_t n,
                                          const unsigned* __restrict__ sin,
                                          unsigned* __restrict__ sout) {
  const float s = slot_scale(sin);
  const float inv = 1.0f / s;
  size_t i = (size_t)blockIdx.x * blockDim.x + threadIdx.x;
  const size_t stride = (size_t)gridDim.x * blockDim.x;
  float m = 0.f;
  for (; i < n; i += stride) {
    float y = fminf(127.f, fmaxf(-128.f, rintf(x[i] * inv))) * s;
    x[i] = y;
    m = fmaxf(m, fabsf(y));
  }
  m = wave_fmax(m);
  if ((threadIdx.x & 31) == 0) atomicMax(sout, __float_as_uint(m));
}

__global__ void qdq_store_kernel(const float* __restrict__ x, size_t n,
                                 const unsigned* __restrict__ slot,
                                 float* __restrict__ y) {
  const float s = slot_scale(slot);
  const float inv = 1.0f / s;
  size_t i = (size_t)blockIdx.x * blockDim.x + threadIdx.x;
  const size_t stride = (size_t)gridDim.x * blockDim.x;
  for (; i < n; i += stride)
    y[i] = fminf(127.f, fmaxf(-128.f, rintf(x[i] * inv))) * s;
}

// ---------------- int8 WMMA GEMM: C = deq(A s8 @ W s8^T) + bias ----------------
// A: [M,K] row-major s8 codes; W: [N,K] row-major s8 codes (B = W^T -> contiguous
// K per output column). Register-blocked: one wave computes a 16x64 strip
// (4 accumulators), reusing the A fragment across 4 back-to-back WMMAs.
// grid = (N/256, M/16), block = 128 (4 waves, 64 cols each).
__global__ void gemm_s8_wmma(const int8_t* __restrict__ A, const int8_t* __restrict__ W,
                             const unsigned* __restrict__ slotA,
                             const unsigned* __restrict__ slotW,
                             const float* __restrict__ bias, float* __restrict__ C,
                             int N, int K) {
  const int lane = threadIdx.x & 31;
  const int wave = threadIdx.x >> 5;
  const int half = lane >> 4;
  const int idx = lane & 15;
  const int m0 = blockIdx.y * 16;
  const int n0 = (blockIdx.x * 4 + wave) * 64;

  const int8_t* arow = A + (size_t)(m0 + idx) * K;
  const int8_t* wrow = W + (size_t)(n0 + idx) * K;  // col-tile nt adds nt*16*K

  v8i acc0 = {0, 0, 0, 0, 0, 0, 0, 0};
  v8i acc1 = {0, 0, 0, 0, 0, 0, 0, 0};
  v8i acc2 = {0, 0, 0, 0, 0, 0, 0, 0};
  v8i acc3 = {0, 0, 0, 0, 0, 0, 0, 0};

  for (int kc = 0; kc < K; kc += 64) {
    __builtin_prefetch(arow + kc + 64, 0, 0);  // global_prefetch_b8
    __builtin_prefetch(wrow + kc + 64, 0, 0);
    v8i af = load_frag(arow + kc, half);
    v8i b0 = load_frag(wrow + kc, half);
    v8i b1 = load_frag(wrow + (size_t)16 * K + kc, half);
    v8i b2 = load_frag(wrow + (size_t)32 * K + kc, half);
    v8i b3 = load_frag(wrow + (size_t)48 * K + kc, half);
    acc0 = __builtin_amdgcn_wmma_i32_16x16x64_iu8(true, af, true, b0, acc0, false, false);
    acc1 = __builtin_amdgcn_wmma_i32_16x16x64_iu8(true, af, true, b1, acc1, false, false);
    acc2 = __builtin_amdgcn_wmma_i32_16x16x64_iu8(true, af, true, b2, acc2, false, false);
    acc3 = __builtin_amdgcn_wmma_i32_16x16x64_iu8(true, af, true, b3, acc3, false, false);
  }

  const float s = slot_scale(slotA) * slot_scale(slotW);
  const int mb = m0 + half * 8;
  v8i accs[4] = {acc0, acc1, acc2, acc3};
#pragma unroll
  for (int nt = 0; nt < 4; ++nt) {
    const int n = n0 + nt * 16 + idx;
    const float bn = bias[n];
#pragma unroll
    for (int r = 0; r < 8; ++r)
      C[(size_t)(mb + r) * N + n] = (float)accs[nt][r] * s + bn;
  }
}

// ---------------- fused attention: scores (WMMA IU8) + softmax + quant_unsigned ----
// grid = (B*H, ceil(S/16)=13), block = 128 (4 waves). projq = int8 codes [B,S,3E].
__global__ void attn_kernel(const int8_t* __restrict__ projq,
                            const unsigned* __restrict__ slot_proj,
                            float* __restrict__ attn_out,      // [B,H,S,S] f32
                            uint8_t* __restrict__ attn_codes)  // [B*H,S,ST] u8 (pad=0)
{
  __shared__ float sc[16][13 * 16];  // 16 rows x 208 score cols
  const int bh = blockIdx.x;
  const int stile = blockIdx.y;
  const int b = bh / HH, h = bh % HH;
  const int s0 = stile * 16;
  const int wave = threadIdx.x >> 5;
  const int lane = threadIdx.x & 31;
  const int half = lane >> 4;
  const int idx = lane & 15;

  const float sp = slot_scale(slot_proj);
  const float qkscale = sp * sp * 0.125f;  // / sqrt(64)

  // ---- scores: A fragment loaded once per wave, reused over 13 col tiles ----
  {
    const int srow = imin(s0 + idx, SS - 1);  // clamp: garbage rows never stored
    const int8_t* qbase = projq + ((size_t)b * SS + srow) * N1 + h * DD;  // sel 0
    v8i af = load_frag(qbase, half);
    for (int t = wave; t < 13; t += 4) {  // wave-uniform trip count, EXEC all-1
      const int tcol = imin(t * 16 + idx, SS - 1);
      const int8_t* kbase = projq + ((size_t)b * SS + tcol) * N1 + EE + h * DD;  // sel 1
      v8i bf = load_frag(kbase, half);
      v8i cz = {0, 0, 0, 0, 0, 0, 0, 0};
      v8i acc = __builtin_amdgcn_wmma_i32_16x16x64_iu8(true, af, true, bf, cz, false, false);
      const int mb = half * 8;
#pragma unroll
      for (int r = 0; r < 8; ++r)
        sc[mb + r][t * 16 + idx] = (float)acc[r] * qkscale;
    }
  }
  __syncthreads();

  // ---- softmax + quant_unsigned per row (one wave per row, 4 rows/wave) ----
  const int rows = imin(16, SS - s0);
  for (int r = wave; r < rows; r += 4) {
    const int s = s0 + r;
    float m = -3.4e38f;
    for (int t = lane; t < SS; t += 32) m = fmaxf(m, sc[r][t]);
    m = wave_fmax(m);
    float sum = 0.f;
    for (int t = lane; t < SS; t += 32) {
      float e = expf(sc[r][t] - m);
      sc[r][t] = e;
      sum += e;
    }
    sum = wave_fsum(sum);
    const float invs = 1.0f / sum;
    const size_t row_out = ((size_t)bh * SS + s) * SS;
    const size_t row_cod = ((size_t)bh * SS + s) * ST;
    for (int t = lane; t < ST; t += 32) {
      uint8_t code = 0;
      if (t < SS) {
        float c = fminf(255.f, fmaxf(0.f, rintf(sc[r][t] * invs * 255.f)));
        code = (uint8_t)c;
        attn_out[row_out + t] = c * (1.0f / 255.0f);
      }
      attn_codes[row_cod + t] = code;  // zero pad t >= 197
    }
  }
}

// ---------------- v transpose: [B,S,3E] codes -> vT s8 [B*H, D, ST] (pad=0) ----
__global__ void vtrans_kernel(const int8_t* __restrict__ projq, int8_t* __restrict__ vT) {
  size_t i = (size_t)blockIdx.x * blockDim.x + threadIdx.x;
  const size_t total = (size_t)BB * HH * DD * ST;
  if (i >= total) return;
  const int t = (int)(i & (ST - 1));
  const size_t rest = i >> 8;
  const int d = (int)(rest & (DD - 1));
  const size_t bh = rest >> 6;
  const int b = (int)(bh / HH), h = (int)(bh % HH);
  int8_t v = 0;
  if (t < SS) v = projq[((size_t)b * SS + t) * N1 + 2 * EE + h * DD + d];
  vT[i] = v;
}

// ---------------- ctx = attn(u8) @ v(s8), IU8 WMMA, K = ST = 256 ----------------
// Register-blocked: one wave computes 16 rows x 64 (full head dim), reusing the
// attention A fragment across 4 WMMAs per K chunk. grid = (13 s-tiles, B*H), block 32.
__global__ void ctx_kernel(const uint8_t* __restrict__ attn_codes,
                           const int8_t* __restrict__ vT,
                           const unsigned* __restrict__ slot_proj,
                           float* __restrict__ ctxf)  // [B,S,E]
{
  const int stile = blockIdx.x;
  const int bh = blockIdx.y;
  const int s0 = stile * 16;
  const int lane = threadIdx.x & 31;
  const int half = lane >> 4;
  const int idx = lane & 15;

  const int srow = imin(s0 + idx, SS - 1);
  const int8_t* abase = (const int8_t*)attn_codes + ((size_t)bh * SS + srow) * ST;
  const int8_t* bbase = vT + ((size_t)bh * DD + idx) * ST;  // d-tile nt adds nt*16*ST

  v8i acc0 = {0, 0, 0, 0, 0, 0, 0, 0};
  v8i acc1 = {0, 0, 0, 0, 0, 0, 0, 0};
  v8i acc2 = {0, 0, 0, 0, 0, 0, 0, 0};
  v8i acc3 = {0, 0, 0, 0, 0, 0, 0, 0};

#pragma unroll
  for (int kc = 0; kc < ST; kc += 64) {
    v8i af = load_frag(abase + kc, half);
    v8i b0 = load_frag(bbase + kc, half);
    v8i b1 = load_frag(bbase + (size_t)16 * ST + kc, half);
    v8i b2 = load_frag(bbase + (size_t)32 * ST + kc, half);
    v8i b3 = load_frag(bbase + (size_t)48 * ST + kc, half);
    // A = unsigned probs codes (0..255), B = signed v codes
    acc0 = __builtin_amdgcn_wmma_i32_16x16x64_iu8(false, af, true, b0, acc0, false, false);
    acc1 = __builtin_amdgcn_wmma_i32_16x16x64_iu8(false, af, true, b1, acc1, false, false);
    acc2 = __builtin_amdgcn_wmma_i32_16x16x64_iu8(false, af, true, b2, acc2, false, false);
    acc3 = __builtin_amdgcn_wmma_i32_16x16x64_iu8(false, af, true, b3, acc3, false, false);
  }

  const float scale = slot_scale(slot_proj) * (1.0f / 255.0f);
  const int b = bh / HH, h = bh % HH;
  const int mb = half * 8;
  v8i accs[4] = {acc0, acc1, acc2, acc3};
#pragma unroll
  for (int nt = 0; nt < 4; ++nt) {
    const int d = nt * 16 + idx;
#pragma unroll
    for (int r = 0; r < 8; ++r) {
      const int s = s0 + mb + r;
      if (s < SS)
        ctxf[((size_t)b * SS + s) * EE + h * DD + d] = (float)accs[nt][r] * scale;
    }
  }
}

// ---------------- workspace layout (bytes, all 256-aligned) ----------------
static constexpr size_t OFF_SLOTS = 0;                          // 16 u32 absmax slots
static constexpr size_t OFF_QX    = 256;
static constexpr size_t SZ_QX     = (size_t)MM * KK;            // 9,682,944
static constexpr size_t OFF_QW1   = OFF_QX + SZ_QX;
static constexpr size_t SZ_QW1    = (size_t)N1 * KK;            // 1,769,472
static constexpr size_t OFF_QW2   = OFF_QW1 + SZ_QW1;
static constexpr size_t SZ_QW2    = (size_t)EE * EE;            // 589,824
static constexpr size_t OFF_PROJQ = OFF_QW2 + SZ_QW2;
static constexpr size_t SZ_PROJQ  = (size_t)MM * N1;            // 29,048,832
static constexpr size_t OFF_PROJF = OFF_PROJQ + SZ_PROJQ;
static constexpr size_t SZ_PROJF  = SZ_PROJQ * 4;               // 116,195,328
// aliases inside PROJF region (proj f32 dead after PROJQ exists):
static constexpr size_t OFF_ATT   = OFF_PROJF;                  // u8, 38,731,776
static constexpr size_t SZ_ATT    = (size_t)BB * HH * SS * ST;
static constexpr size_t OFF_VT    = OFF_ATT + SZ_ATT;           // s8, 12,582,912
static constexpr size_t SZ_VT     = (size_t)BB * HH * DD * ST;
static constexpr size_t OFF_CTXF  = OFF_VT + SZ_VT;             // f32, 38,731,776
static constexpr size_t OFF_OUTF  = OFF_PROJF;                  // reuse after ctx GEMM
static constexpr size_t OFF_QCTX  = OFF_QX;                     // reuse qx after GEMM1
static constexpr size_t WS_NEEDED = OFF_PROJF + SZ_PROJF;       // ~157 MB

extern "C" void kernel_launch(void* const* d_in, const int* in_sizes, int n_in,
                              void* d_out, int out_size, void* d_ws, size_t ws_size,
                              hipStream_t stream) {
  if (ws_size < WS_NEEDED) return;

  const float* input  = (const float*)d_in[0];
  const float* w1     = (const float*)d_in[1];
  const float* b1     = (const float*)d_in[2];
  const float* w2     = (const float*)d_in[3];
  const float* b2     = (const float*)d_in[4];

  char* ws = (char*)d_ws;
  unsigned* slots = (unsigned*)(ws + OFF_SLOTS);
  int8_t*  qx    = (int8_t*)(ws + OFF_QX);
  int8_t*  qw1   = (int8_t*)(ws + OFF_QW1);
  int8_t*  qw2   = (int8_t*)(ws + OFF_QW2);
  int8_t*  projq = (int8_t*)(ws + OFF_PROJQ);
  float*   projf = (float*)(ws + OFF_PROJF);
  uint8_t* attnc = (uint8_t*)(ws + OFF_ATT);
  int8_t*  vT    = (int8_t*)(ws + OFF_VT);
  float*   ctxf  = (float*)(ws + OFF_CTXF);
  float*   outf  = (float*)(ws + OFF_OUTF);
  int8_t*  qctx  = (int8_t*)(ws + OFF_QCTX);

  float* outp    = (float*)d_out;                  // [B,S,E]
  float* attnout = outp + (size_t)MM * EE;         // [B,H,S,S]

  const size_t n_in0 = (size_t)MM * KK;
  const size_t n_w1  = (size_t)N1 * KK;
  const size_t n_w2  = (size_t)EE * EE;
  const size_t n_pr  = (size_t)MM * N1;
  const size_t n_ctx = (size_t)MM * EE;

  auto blocks = [](size_t n) { int b = (int)((n + 255) / 256); return b > 4096 ? 4096 : b; };

  // slots: 0=|x| 1=|w1| 2=|w2| 3=|proj| 4=|ctx| 5=|ctx_q| 6=|out|
  zero_slots_kernel<<<1, 32, 0, stream>>>(slots);

  // [1] quantize activations and weights
  absmax_kernel<<<blocks(n_in0), 256, 0, stream>>>(input, n_in0, slots + 0);
  absmax_kernel<<<blocks(n_w1), 256, 0, stream>>>(w1, n_w1, slots + 1);
  absmax_kernel<<<blocks(n_w2), 256, 0, stream>>>(w2, n_w2, slots + 2);
  quantize_s8_kernel<<<blocks(n_in0), 256, 0, stream>>>(input, n_in0, slots + 0, qx);
  quantize_s8_kernel<<<blocks(n_w1), 256, 0, stream>>>(w1, n_w1, slots + 1, qw1);
  quantize_s8_kernel<<<blocks(n_w2), 256, 0, stream>>>(w2, n_w2, slots + 2, qw2);

  // in_proj int8 WMMA GEMM: [12608,2304] = [12608,768] @ [2304,768]^T
  gemm_s8_wmma<<<dim3(N1 / 256, MM / 16), 128, 0, stream>>>(
      qx, qw1, slots + 0, slots + 1, b1, projf, N1, KK);

  // quant_sym(proj) -> q/k/v int8 codes
  absmax_kernel<<<blocks(n_pr), 256, 0, stream>>>(projf, n_pr, slots + 3);
  quantize_s8_kernel<<<blocks(n_pr), 256, 0, stream>>>(projf, n_pr, slots + 3, projq);

  // [2,3] fused scores + softmax + quant_unsigned (writes attn_map to d_out)
  attn_kernel<<<dim3(BB * HH, (SS + 15) / 16), 128, 0, stream>>>(
      projq, slots + 3, attnout, attnc);

  // [4] ctx = attn @ v via IU8 WMMA (v pre-transposed for contiguous K)
  vtrans_kernel<<<(int)(SZ_VT / 256), 256, 0, stream>>>(projq, vT);
  ctx_kernel<<<dim3((SS + 15) / 16, BB * HH), 32, 0, stream>>>(
      attnc, vT, slots + 3, ctxf);

  // ctx = quant_sym(ctx); then quant_sym again feeding out_proj
  absmax_kernel<<<blocks(n_ctx), 256, 0, stream>>>(ctxf, n_ctx, slots + 4);
  qdq_inplace_absmax_kernel<<<blocks(n_ctx), 256, 0, stream>>>(ctxf, n_ctx, slots + 4, slots + 5);
  quantize_s8_kernel<<<blocks(n_ctx), 256, 0, stream>>>(ctxf, n_ctx, slots + 5, qctx);

  // [5] out_proj int8 WMMA GEMM + final quant_sym into d_out
  gemm_s8_wmma<<<dim3(EE / 256, MM / 16), 128, 0, stream>>>(
      qctx, qw2, slots + 5, slots + 2, b2, outf, EE, KK);
  absmax_kernel<<<blocks(n_ctx), 256, 0, stream>>>(outf, n_ctx, slots + 6);
  qdq_store_kernel<<<blocks(n_ctx), 256, 0, stream>>>(outf, n_ctx, slots + 6, outp);

  (void)in_sizes; (void)n_in; (void)out_size;
}
